// AttentionBlock_3719441678377
// MI455X (gfx1250) — matrix operand order repaired and checked by reading, not compiled, not run on hardware
//
#include <hip/hip_runtime.h>

typedef unsigned short u16t;
typedef __attribute__((ext_vector_type(16))) __bf16 v16bf;
typedef __attribute__((ext_vector_type(8)))  float  v8f;

union BF16Frag { v16bf v; uint4 q[2]; u16t e[16]; };

#define CB 1024
#define SB 2048
#define HB 16
#define DB 64
#define BB 2
#define QT 16
#define KT 32

__device__ __forceinline__ u16t f2bf(float f) {
  union { float f; unsigned u; } x; x.f = f;
  unsigned u = x.u;
  return (u16t)((u + 0x7FFFu + ((u >> 16) & 1u)) >> 16);  // RNE
}

__device__ __forceinline__ v8f wmma_bf16(v16bf a, v16bf b, v8f c) {
  // v_wmma_f32_16x16x32_bf16
  return __builtin_amdgcn_wmma_f32_16x16x32_bf16(false, a, false, b, (short)0, c,
                                                 false, false);
}

__device__ __forceinline__ float grp_max(float v) {  // reduce within 16-lane group
  v = fmaxf(v, __shfl_xor(v, 1, 32));
  v = fmaxf(v, __shfl_xor(v, 2, 32));
  v = fmaxf(v, __shfl_xor(v, 4, 32));
  v = fmaxf(v, __shfl_xor(v, 8, 32));
  return v;
}
__device__ __forceinline__ float grp_sum(float v) {
  v += __shfl_xor(v, 1, 32);
  v += __shfl_xor(v, 2, 32);
  v += __shfl_xor(v, 4, 32);
  v += __shfl_xor(v, 8, 32);
  return v;
}

// Load the 4 QK^T B-fragments (2 k-subtiles x 2 d-chunks) for k-tile kk.
__device__ __forceinline__ void load_kfrags(BF16Frag kb[2][2],
                                            const u16t* __restrict__ Kbase,
                                            int kk, int lm, int lg) {
#pragma unroll
  for (int dc = 0; dc < 2; ++dc) {
    const u16t* p0 = Kbase + (size_t)(kk + lm) * DB + dc * 32 + lg * 8;
    const u16t* p1 = Kbase + (size_t)(kk + 16 + lm) * DB + dc * 32 + lg * 8;
    kb[0][dc].q[0] = *(const uint4*)p0;
    kb[0][dc].q[1] = *(const uint4*)(p0 + 16);
    kb[1][dc].q[0] = *(const uint4*)p1;
    kb[1][dc].q[1] = *(const uint4*)(p1 + 16);
  }
}

// ---------------------------------------------------------------- converts
__global__ void cvt_f32_to_bf16(const float* __restrict__ x, u16t* __restrict__ y,
                                int n) {
  int i = blockIdx.x * blockDim.x + threadIdx.x;
  int stride = gridDim.x * blockDim.x;
  for (; i < n; i += stride) y[i] = f2bf(x[i]);
}

// x[rows][cols] f32 -> y[cols][rows] bf16 (scaled). Gives N-major weights so
// WMMA B-fragments are pairs of contiguous 16B loads along k.
__global__ void transpose_cvt(const float* __restrict__ x, u16t* __restrict__ y,
                              int rows, int cols, float scale) {
  int i = blockIdx.x * blockDim.x + threadIdx.x;
  int stride = gridDim.x * blockDim.x;
  int n = rows * cols;
  for (; i < n; i += stride) {
    int r = i / cols, c = i - r * cols;
    y[(size_t)c * rows + r] = f2bf(x[i] * scale);
  }
}

// --------------------------------------------------- generic bf16 WMMA GEMM
// Each wave computes a 16x64 tile of out = A[MxK] * Bt^T (Bt stored [N][K]).
// Manually double-buffered over k (two disjoint fragment sets, no copies):
// each half consumes fragments loaded a full 64-wide step earlier, then
// immediately re-issues that buffer's loads, so WMMAs overlap L2 latency.
// mode 0: store bf16 as [b,h,s,d]   (Q, K projections)
// mode 1: store bf16 as [b,h,d,s]   (V projection, pre-transposed for PV WMMA)
// mode 2: store f32 row-major       (final output projection)
__global__ __launch_bounds__(256) void gemm_bf16_16x64(
    const u16t* __restrict__ A, const u16t* __restrict__ Bt, int M, int N, int K,
    int mode, int bS, u16t* __restrict__ outb, float* __restrict__ outf) {
  const int wid = blockIdx.x * (blockDim.x >> 5) + (threadIdx.x >> 5);
  const int ntiles = N >> 6;
  const int mt = wid / ntiles;
  const int nt = wid - mt * ntiles;
  if (mt * 16 >= M) return;
  const int lane = threadIdx.x & 31;
  const int lm = lane & 15;
  const int lg = lane >> 4;
  const int khA = lg ? 8 : 0;

  v8f z = {};
  v8f acc[4] = {z, z, z, z};
  const u16t* Arow = A + (size_t)(mt * 16 + lm) * K + khA;
  const u16t* Brow[4];
#pragma unroll
  for (int t = 0; t < 4; ++t)
    Brow[t] = Bt + (size_t)(nt * 64 + t * 16 + lm) * K + lg * 8;

  // Prologue: buffer 0 <- k=0, buffer 1 <- k=32.
  BF16Frag ua0, ub0[4], ua1, ub1[4];
  ua0.q[0] = *(const uint4*)(Arow);
  ua0.q[1] = *(const uint4*)(Arow + 16);
#pragma unroll
  for (int t = 0; t < 4; ++t) {
    ub0[t].q[0] = *(const uint4*)(Brow[t]);
    ub0[t].q[1] = *(const uint4*)(Brow[t] + 16);
  }
  ua1.q[0] = *(const uint4*)(Arow + 32);
  ua1.q[1] = *(const uint4*)(Arow + 48);
#pragma unroll
  for (int t = 0; t < 4; ++t) {
    ub1[t].q[0] = *(const uint4*)(Brow[t] + 32);
    ub1[t].q[1] = *(const uint4*)(Brow[t] + 48);
  }

  for (int k0 = 0; k0 < K; k0 += 64) {
    // ---- half 0: consume buffer 0 (k0), refill it for k0+64 ----
#pragma unroll
    for (int t = 0; t < 4; ++t) acc[t] = wmma_bf16(ua0.v, ub0[t].v, acc[t]);
    int ka = k0 + 64;
    if (ka >= K) ka = 0;  // wrap: harmless dummy prefetch at the tail
    ua0.q[0] = *(const uint4*)(Arow + ka);
    ua0.q[1] = *(const uint4*)(Arow + ka + 16);
#pragma unroll
    for (int t = 0; t < 4; ++t) {
      ub0[t].q[0] = *(const uint4*)(Brow[t] + ka);
      ub0[t].q[1] = *(const uint4*)(Brow[t] + ka + 16);
    }
    // ---- half 1: consume buffer 1 (k0+32), refill it for k0+96 ----
#pragma unroll
    for (int t = 0; t < 4; ++t) acc[t] = wmma_bf16(ua1.v, ub1[t].v, acc[t]);
    int kbn = k0 + 96;
    if (kbn >= K) kbn = 32;  // wrap: harmless dummy prefetch at the tail
    ua1.q[0] = *(const uint4*)(Arow + kbn);
    ua1.q[1] = *(const uint4*)(Arow + kbn + 16);
#pragma unroll
    for (int t = 0; t < 4; ++t) {
      ub1[t].q[0] = *(const uint4*)(Brow[t] + kbn);
      ub1[t].q[1] = *(const uint4*)(Brow[t] + kbn + 16);
    }
  }

#pragma unroll
  for (int t = 0; t < 4; ++t) {
#pragma unroll
    for (int r = 0; r < 8; ++r) {
      int row = mt * 16 + r + 8 * lg;  // C/D layout: M = r + 8*group
      int col = nt * 64 + t * 16 + lm; // N = lane%16
      float v = acc[t][r];
      if (mode == 2) {
        outf[(size_t)row * N + col] = v;
      } else {
        int bb = row / bS, s = row - bb * bS;
        int h = col >> 6, d = col & 63;
        if (mode == 0)
          outb[(((size_t)bb * HB + h) * bS + s) * DB + d] = f2bf(v);
        else
          outb[(((size_t)bb * HB + h) * DB + d) * bS + s] = f2bf(v);
      }
    }
  }
}

// ----------------------------------------------- fused talking-heads attention
// One workgroup = one (batch, 16-row q tile); 16 waves, wave w owns head w.
// Two passes over K (all of K/V is 16 MB bf16 -> L2 resident, re-reads cheap).
// K fragments for step k+KT and V fragments for step k are issued before the
// LDS-exchange/softmax phase, which hides their L2 latency.
__global__ __launch_bounds__(512) void attn_talking_heads(
    const u16t* __restrict__ Qg,   // [b,h,s,d] bf16, pre-scaled by 1/sqrt(D)
    const u16t* __restrict__ Kg,   // [b,h,s,d] bf16
    const u16t* __restrict__ Vt,   // [b,h,d,s] bf16
    const float* __restrict__ Wth1, const float* __restrict__ Wth2,
    u16t* __restrict__ AO) {       // [b,s,h*64+d] bf16
  __shared__ float exch[HB][2][32][8];  // lane-major 16x16 tiles, 32 KB

  const int wave = threadIdx.x >> 5;
  const int lane = threadIdx.x & 31;
  const int lm = lane & 15;
  const int lg = lane >> 4;
  const int b = blockIdx.x / (SB / QT);
  const int q0 = (blockIdx.x % (SB / QT)) * QT;
  const int khA = lg ? 8 : 0;

  const u16t* Qbase = Qg + (((size_t)b * HB + wave) * SB + q0) * DB;
  const u16t* Kbase = Kg + (((size_t)b * HB + wave) * SB) * DB;
  const u16t* Vbase = Vt + (((size_t)b * HB + wave) * DB) * SB;

  // Q fragments for this head, kept in registers for the whole kernel.
  v16bf qa[2];
#pragma unroll
  for (int f = 0; f < 2; ++f) {
    BF16Frag t;
    const u16t* p = Qbase + (size_t)lm * DB + f * 32 + khA;
    t.q[0] = *(const uint4*)(p);
    t.q[1] = *(const uint4*)(p + 16);
    qa[f] = t.v;
  }

  float w1[HB], w2[HB];
#pragma unroll
  for (int h = 0; h < HB; ++h) {
    w1[h] = Wth1[h * HB + wave];
    w2[h] = Wth2[h * HB + wave];
  }

  float mrun[8], zrun[8];
#pragma unroll
  for (int r = 0; r < 8; ++r) { mrun[r] = -3.0e38f; zrun[r] = 0.0f; }

  BF16Frag kb[2][2];
  load_kfrags(kb, Kbase, 0, lm, lg);  // prologue for pass 1

  // -------- pass 1: exact per-head softmax stats (max, Z) after Wth1 mix ----
  for (int k0 = 0; k0 < SB; k0 += KT) {
    v8f acc0 = {}, acc1 = {};
#pragma unroll
    for (int dc = 0; dc < 2; ++dc) {
      acc0 = wmma_bf16(qa[dc], kb[0][dc].v, acc0);
      acc1 = wmma_bf16(qa[dc], kb[1][dc].v, acc1);
    }
    int kn = k0 + KT;
    if (kn >= SB) kn = 0;  // wrap: preloads pass 2's first k-tile
    load_kfrags(kb, Kbase, kn, lm, lg);

    __syncthreads();
#pragma unroll
    for (int r = 0; r < 8; ++r) {
      exch[wave][0][lane][r] = acc0[r];
      exch[wave][1][lane][r] = acc1[r];
    }
    __syncthreads();
    float a0[8] = {}, a1[8] = {};
#pragma unroll
    for (int h = 0; h < HB; ++h) {
      float w = w1[h];
#pragma unroll
      for (int r = 0; r < 8; ++r) {
        a0[r] = fmaf(w, exch[h][0][lane][r], a0[r]);
        a1[r] = fmaf(w, exch[h][1][lane][r], a1[r]);
      }
    }
#pragma unroll
    for (int r = 0; r < 8; ++r) {
      float rv = grp_max(fmaxf(a0[r], a1[r]));
      float mnew = fmaxf(mrun[r], rv);
      float sc = __expf(mrun[r] - mnew);
      float s = grp_sum(__expf(a0[r] - mnew) + __expf(a1[r] - mnew));
      zrun[r] = zrun[r] * sc + s;
      mrun[r] = mnew;
    }
  }

  float zinv[8];
#pragma unroll
  for (int r = 0; r < 8; ++r) zinv[r] = 1.0f / zrun[r];

  v8f zv = {};
  v8f oacc[4] = {zv, zv, zv, zv};

  // -------- pass 2: recompute, mix1 -> exp/Z -> mix2 -> P@V -----------------
  // kb already holds the k0 = 0 fragments (wrapped prefetch from pass 1).
  for (int k0 = 0; k0 < SB; k0 += KT) {
    v8f acc0 = {}, acc1 = {};
#pragma unroll
    for (int dc = 0; dc < 2; ++dc) {
      acc0 = wmma_bf16(qa[dc], kb[0][dc].v, acc0);
      acc1 = wmma_bf16(qa[dc], kb[1][dc].v, acc1);
    }
    // Issue V fragments for this step now; consumed after the exchange phase.
    BF16Frag vb[4];
#pragma unroll
    for (int t = 0; t < 4; ++t) {
      const u16t* pv = Vbase + (size_t)(t * 16 + lm) * SB + k0 + lg * 8;
      vb[t].q[0] = *(const uint4*)pv;
      vb[t].q[1] = *(const uint4*)(pv + 16);
    }
    int kn = k0 + KT;
    if (kn >= SB) kn = 0;  // final wrap load is a harmless dummy
    load_kfrags(kb, Kbase, kn, lm, lg);

    __syncthreads();  // prior iteration's LDS reads complete
#pragma unroll
    for (int r = 0; r < 8; ++r) {
      exch[wave][0][lane][r] = acc0[r];
      exch[wave][1][lane][r] = acc1[r];
    }
    __syncthreads();
    float a0[8] = {}, a1[8] = {};
#pragma unroll
    for (int h = 0; h < HB; ++h) {
      float w = w1[h];
#pragma unroll
      for (int r = 0; r < 8; ++r) {
        a0[r] = fmaf(w, exch[h][0][lane][r], a0[r]);
        a1[r] = fmaf(w, exch[h][1][lane][r], a1[r]);
      }
    }
    float p0[8], p1[8];
#pragma unroll
    for (int r = 0; r < 8; ++r) {
      p0[r] = __expf(a0[r] - mrun[r]) * zinv[r];
      p1[r] = __expf(a1[r] - mrun[r]) * zinv[r];
    }
    __syncthreads();
#pragma unroll
    for (int r = 0; r < 8; ++r) {
      exch[wave][0][lane][r] = p0[r];
      exch[wave][1][lane][r] = p1[r];
    }
    __syncthreads();
    float c0[8] = {}, c1[8] = {};
#pragma unroll
    for (int h = 0; h < HB; ++h) {
      float w = w2[h];
#pragma unroll
      for (int r = 0; r < 8; ++r) {
        c0[r] = fmaf(w, exch[h][0][lane][r], c0[r]);
        c1[r] = fmaf(w, exch[h][1][lane][r], c1[r]);
      }
    }
    __syncthreads();
#pragma unroll
    for (int r = 0; r < 8; ++r) {
      exch[wave][0][lane][r] = c0[r];
      exch[wave][1][lane][r] = c1[r];
    }
    __syncthreads();
    // Gather mixed probabilities as a bf16 A-fragment (16x32 over k).
    BF16Frag af;
#pragma unroll
    for (int e = 0; e < 8; ++e) {
      int n = khA + e;                      // column within 16-wide k subtile
      int sl = n + ((lm >> 3) << 4);        // source lane in acc layout
      int rr = lm & 7;                      // source VGPR index
      af.e[e]     = f2bf(exch[wave][0][sl][rr]);
      af.e[8 + e] = f2bf(exch[wave][1][sl][rr]);
    }
    // P @ V with the pre-issued V fragments.
#pragma unroll
    for (int t = 0; t < 4; ++t) oacc[t] = wmma_bf16(af.v, vb[t].v, oacc[t]);
  }

  // Store attention output as [b, s, h*64+d] bf16, ready for the Wo GEMM.
#pragma unroll
  for (int t = 0; t < 4; ++t) {
#pragma unroll
    for (int r = 0; r < 8; ++r) {
      int row = q0 + r + 8 * lg;
      int col = wave * DB + t * 16 + lm;
      AO[((size_t)b * SB + row) * (HB * DB) + col] = f2bf(oacc[t][r]);
    }
  }
}

// ---------------------------------------------------------------- launcher
extern "C" void kernel_launch(void* const* d_in, const int* in_sizes, int n_in,
                              void* d_out, int out_size, void* d_ws, size_t ws_size,
                              hipStream_t stream) {
  (void)in_sizes; (void)n_in; (void)out_size;
  const float* inq  = (const float*)d_in[0];
  const float* inkv = (const float*)d_in[1];
  // d_in[2] = is_training (unused in inference path)
  const float* Wq   = (const float*)d_in[3];
  const float* Wk   = (const float*)d_in[4];
  const float* Wv   = (const float*)d_in[5];
  const float* Wth1 = (const float*)d_in[6];
  const float* Wth2 = (const float*)d_in[7];
  const float* Wo   = (const float*)d_in[8];
  float* out = (float*)d_out;

  const size_t MROWS = (size_t)BB * SB;       // 4096
  const size_t ACT = MROWS * CB;              // 4M elements
  const size_t WEL = (size_t)CB * (HB * DB);  // 1M elements

  char* p = (char*)d_ws;
  u16t* Xq  = (u16t*)p; p += ACT * 2;
  u16t* Xkv = (u16t*)p; p += ACT * 2;
  u16t* Wqt = (u16t*)p; p += WEL * 2;
  u16t* Wkt = (u16t*)p; p += WEL * 2;
  u16t* Wvt = (u16t*)p; p += WEL * 2;
  u16t* Wot = (u16t*)p; p += WEL * 2;
  u16t* Qg  = (u16t*)p; p += ACT * 2;
  u16t* Kg  = (u16t*)p; p += ACT * 2;
  u16t* Vtg = (u16t*)p; p += ACT * 2;
  u16t* AO  = (u16t*)p; p += ACT * 2;
  if ((size_t)(p - (char*)d_ws) > ws_size) return;  // ~56 MB needed

  cvt_f32_to_bf16<<<1024, 256, 0, stream>>>(inq, Xq, (int)ACT);
  cvt_f32_to_bf16<<<1024, 256, 0, stream>>>(inkv, Xkv, (int)ACT);
  // Weights -> bf16, N-major; 1/sqrt(D)=0.125 folded into Wq.
  transpose_cvt<<<1024, 256, 0, stream>>>(Wq, Wqt, CB, HB * DB, 0.125f);
  transpose_cvt<<<1024, 256, 0, stream>>>(Wk, Wkt, CB, HB * DB, 1.0f);
  transpose_cvt<<<1024, 256, 0, stream>>>(Wv, Wvt, CB, HB * DB, 1.0f);
  transpose_cvt<<<1024, 256, 0, stream>>>(Wo, Wot, HB * DB, CB, 1.0f);

  const int gemm_waves = (int)(MROWS / 16) * ((HB * DB) / 64);  // 4096
  const int gemm_blocks = gemm_waves / 8;                       // 512
  gemm_bf16_16x64<<<gemm_blocks, 256, 0, stream>>>(Xq, Wqt, (int)MROWS, HB * DB,
                                                   CB, 0, SB, Qg, nullptr);
  gemm_bf16_16x64<<<gemm_blocks, 256, 0, stream>>>(Xkv, Wkt, (int)MROWS, HB * DB,
                                                   CB, 0, SB, Kg, nullptr);
  gemm_bf16_16x64<<<gemm_blocks, 256, 0, stream>>>(Xkv, Wvt, (int)MROWS, HB * DB,
                                                   CB, 1, SB, Vtg, nullptr);

  attn_talking_heads<<<BB * (SB / QT), 512, 0, stream>>>(Qg, Kg, Vtg, Wth1, Wth2,
                                                         AO);

  gemm_bf16_16x64<<<gemm_blocks, 256, 0, stream>>>(AO, Wot, (int)MROWS, CB,
                                                   HB * DB, 2, SB, nullptr, out);
}
